// Encoder_60911226192171
// MI455X (gfx1250) — matrix operand order repaired and checked by reading, not compiled, not run on hardware
//
#include <hip/hip_runtime.h>
#include <hip/hip_bf16.h>

typedef __attribute__((ext_vector_type(16))) __bf16 v16bf;
typedef __attribute__((ext_vector_type(8)))  __bf16 v8bf;
typedef __attribute__((ext_vector_type(8)))  float  v8f;

// ---------------------------------------------------------------------------
// Streaming f32 -> bf16 conversion (optionally fused ReLU).
// 8 elements per thread: 2x global_load_b128 in, 4x v_cvt_pk_bf16_f32,
// 1x global_store_b128 out.
// ---------------------------------------------------------------------------
template <int RELU>
__global__ __launch_bounds__(256)
void cvt_f32_bf16(const float* __restrict__ in, __bf16* __restrict__ out, int n) {
    const int i = (blockIdx.x * 256 + threadIdx.x) * 8;
    if (i >= n) return;
    const float4 f0 = *(const float4*)(in + i);
    const float4 f1 = *(const float4*)(in + i + 4);
    float av[8] = {f0.x, f0.y, f0.z, f0.w, f1.x, f1.y, f1.z, f1.w};
    v8bf o;
    #pragma unroll
    for (int j = 0; j < 8; ++j) {
        float t = av[j];
        if (RELU) t = fmaxf(t, 0.0f);
        o[j] = (__bf16)t;
    }
    *(v8bf*)(out + i) = o;
}

// ---------------------------------------------------------------------------
// Dense GEMM: C[M,N] = A[M,K=256](bf16) * W[K=256,N](f32->bf16 in LDS).
// Block = 256 threads = 8 waves. Block tile = 128 (M) x 64 (N).
// Wave w computes rows [blk*128 + w*16, +16) x cols [blockIdx.y*64, +64)
// via 4 accumulators of v_wmma_f32_16x16x32_bf16, K-loop of 8 steps.
// W panel (256K x 64N) is pre-swizzled into LDS in the per-lane WMMA
// B-fragment layout so each fragment is one 32B LDS vector load.
// A is pre-converted bf16: each fragment = two contiguous 16B b128 loads.
// ---------------------------------------------------------------------------
__global__ __launch_bounds__(256)
void gemm_bf16_wmma(const __bf16* __restrict__ A, const float* __restrict__ W,
                    float* __restrict__ C, int M, int N) {
    // LDS: [kstep(8)][ntile(4)][lane(32)][elem(16)] bf16 = 32 KB
    __shared__ __bf16 sW[8 * 4 * 32 * 16];

    const int tid   = threadIdx.x;
    const int nbase = blockIdx.y * 64;

    // ---- Pack weight panel into LDS in WMMA-B fragment layout -------------
    // B fragment is 32x16 (KxN): lane (l&15) = column, lane>>4 selects the
    // K ranges {0-7,16-23} vs {8-15,24-31} (same order as A).
    for (int idx = tid; idx < 8 * 4 * 32 * 16; idx += 256) {
        const int j     = idx & 15;
        const int lane  = (idx >> 4) & 31;
        const int ntile = (idx >> 9) & 3;
        const int kstep = idx >> 11;
        const int col   = nbase + ntile * 16 + (lane & 15);
        const int kk    = kstep * 32 + ((lane >> 4) << 3) + (j & 7) + ((j >> 3) << 4);
        sW[idx] = (__bf16)W[kk * N + col];
    }
    __syncthreads();

    const int wave  = tid >> 5;
    const int lane  = tid & 31;
    const int mwave = blockIdx.x * 128 + wave * 16;

    // A layout (16-bit 16x32): lanes 0-15 & 16-31 both hold rows M=0..15;
    // lane half selects K ranges {0-7,16-23} vs {8-15,24-31}.
    const int arow  = mwave + (lane & 15);
    const int arowc = arow < M ? arow : (M - 1);          // clamp, keep EXEC full
    const __bf16* Abase = A + (size_t)arowc * 256;
    const int khalf8 = (lane >> 4) << 3;                   // 0 or 8

    v8f acc0 = {}, acc1 = {}, acc2 = {}, acc3 = {};

    #pragma unroll
    for (int kstep = 0; kstep < 8; ++kstep) {
        const int kb = kstep * 32 + khalf8;
        const v8bf alo = *(const v8bf*)(Abase + kb);        // K = kb..kb+7
        const v8bf ahi = *(const v8bf*)(Abase + kb + 16);   // K = kb+16..kb+23
        const v16bf a = __builtin_shufflevector(alo, ahi,
                             0, 1, 2, 3, 4, 5, 6, 7, 8, 9, 10, 11, 12, 13, 14, 15);

        const v16bf b0 = *(const v16bf*)&sW[((kstep * 4 + 0) * 32 + lane) * 16];
        const v16bf b1 = *(const v16bf*)&sW[((kstep * 4 + 1) * 32 + lane) * 16];
        const v16bf b2 = *(const v16bf*)&sW[((kstep * 4 + 2) * 32 + lane) * 16];
        const v16bf b3 = *(const v16bf*)&sW[((kstep * 4 + 3) * 32 + lane) * 16];

        acc0 = __builtin_amdgcn_wmma_f32_16x16x32_bf16(false, a, false, b0,
                                                       (short)0, acc0, false, false);
        acc1 = __builtin_amdgcn_wmma_f32_16x16x32_bf16(false, a, false, b1,
                                                       (short)0, acc1, false, false);
        acc2 = __builtin_amdgcn_wmma_f32_16x16x32_bf16(false, a, false, b2,
                                                       (short)0, acc2, false, false);
        acc3 = __builtin_amdgcn_wmma_f32_16x16x32_bf16(false, a, false, b3,
                                                       (short)0, acc3, false, false);
    }

    // C/D layout: VGPR i -> row i (lanes 0-15) or i+8 (lanes 16-31); col = lane&15
    const int rofs  = (lane >> 4) << 3;
    const int cbase = nbase + (lane & 15);
    v8f accs[4] = {acc0, acc1, acc2, acc3};
    #pragma unroll
    for (int t = 0; t < 4; ++t) {
        const int col = cbase + t * 16;
        #pragma unroll
        for (int i = 0; i < 8; ++i) {
            const int r = mwave + rofs + i;
            if (r < M) C[(size_t)r * N + col] = accs[t][i];
        }
    }
}

// ---------------------------------------------------------------------------
// Device-scope, no-return fp32 atomic add (CDNA5 global_atomic_add_f32).
// No-return form tracks STOREcnt only; kernel end performs the final wait.
// ---------------------------------------------------------------------------
__device__ __forceinline__ void atomic_add_f32_dev(float* addr, float val) {
    asm volatile("global_atomic_add_f32 %0, %1, off scope:SCOPE_DEV"
                 :
                 : "v"(addr), "v"(val)
                 : "memory");
}

// ---------------------------------------------------------------------------
// COO SpMM scatter: y[row[e], :] += val[e] * x[col[e], :]
// One wave per edge; each lane owns FPL consecutive features.
// FPL = 8 -> F = 256, FPL = 4 -> F = 128.
// Gathers hit L2 (x panels fit in the 192MB L2); scatter-adds run in the L2
// atomic units.
// ---------------------------------------------------------------------------
template <int FPL>
__global__ __launch_bounds__(256)
void spmm_scatter(const int* __restrict__ erow, const int* __restrict__ ecol,
                  const float* __restrict__ eval, const float* __restrict__ x,
                  float* __restrict__ y, int E) {
    const int gid  = blockIdx.x * blockDim.x + threadIdx.x;
    const int edge = gid >> 5;
    const int lane = gid & 31;
    if (edge >= E) return;

    const int   r = erow[edge];
    const int   c = ecol[edge];
    const float v = eval[edge];

    constexpr int F = FPL * 32;
    const float* xp = x + (size_t)c * F + lane * FPL;
    float*       yp = y + (size_t)r * F + lane * FPL;

    float xv[FPL];
    #pragma unroll
    for (int i = 0; i < FPL; ++i) xv[i] = xp[i];   // merges into b128 loads

    #pragma unroll
    for (int i = 0; i < FPL; ++i) {
        atomic_add_f32_dev(yp + i, v * xv[i]);
    }
}

// ---------------------------------------------------------------------------
extern "C" void kernel_launch(void* const* d_in, const int* in_sizes, int n_in,
                              void* d_out, int out_size, void* d_ws, size_t ws_size,
                              hipStream_t stream) {
    const float* Y    = (const float*)d_in[0];
    const int*   erow = (const int*)d_in[1];
    const int*   ecol = (const int*)d_in[2];
    const float* eval = (const float*)d_in[3];
    const float* W1   = (const float*)d_in[4];
    const float* W2   = (const float*)d_in[5];

    const int M = in_sizes[0] / 256;   // 50000 nodes
    const int E = in_sizes[1];         // 800000 edges

    // Workspace layout (lifetime-aliased, 128 MB total for M=50000):
    //  bufA (M*256*2B = 25.6MB): Ybf16 during layer 1, then H2 [M,128] f32.
    //  bufB (M*256*4B = 51.2MB): H0 [M,256] f32, then hbf16 [M,256] bf16.
    //  bufC (M*256*4B = 51.2MB): hb [M,256] f32 atomic aggregate.
    char* ws = (char*)d_ws;
    __bf16* Ybf = (__bf16*)ws;                                  // bufA
    float*  H2  = (float*)ws;                                   // bufA (later)
    float*  H0  = (float*)(ws + (size_t)M * 256 * 2);           // bufB
    __bf16* hbf = (__bf16*)(ws + (size_t)M * 256 * 2);          // bufB (later)
    float*  hb  = (float*)(ws + (size_t)M * 256 * 2 + (size_t)M * 256 * 4); // bufC
    float*  out = (float*)d_out;                                // [M,128]

    // Zero the aggregation targets (graph-capture-safe).
    hipMemsetAsync(hb, 0, (size_t)M * 256 * sizeof(float), stream);
    hipMemsetAsync(out, 0, (size_t)out_size * sizeof(float), stream);

    const int nY      = M * 256;
    const int cblocks = (nY / 8 + 255) / 256;
    const int mblocks = (M + 127) / 128;
    const int eblocks = (E * 32 + 255) / 256;

    // Layer 1: Ybf = bf16(Y); H0 = Ybf @ W1
    cvt_f32_bf16<0><<<cblocks, 256, 0, stream>>>(Y, Ybf, nY);
    gemm_bf16_wmma<<<dim3(mblocks, 4), 256, 0, stream>>>(Ybf, W1, H0, M, 256);
    // hb = spmm(H0)
    spmm_scatter<8><<<eblocks, 256, 0, stream>>>(erow, ecol, eval, H0, hb, E);
    // Layer 2: hbf = bf16(relu(hb)); H2 = hbf @ W2
    cvt_f32_bf16<1><<<cblocks, 256, 0, stream>>>(hb, hbf, nY);
    gemm_bf16_wmma<<<dim3(mblocks, 2), 256, 0, stream>>>(hbf, W2, H2, M, 128);
    // out = spmm(H2)
    spmm_scatter<4><<<eblocks, 256, 0, stream>>>(erow, ecol, eval, H2, out, E);
}
